// ada_filter_50379966382597
// MI455X (gfx1250) — matrix operand order repaired and checked by reading, not compiled, not run on hardware
//
#include <hip/hip_runtime.h>
#include <stdint.h>

#define N_NODES 50000
#define IN_C    512
#define HID_C   256
#define OUT_C   64
#define P_HOPS  10

typedef __attribute__((ext_vector_type(16))) __bf16 v16bf;
typedef __attribute__((ext_vector_type(8)))  float  v8f;

// trivial union for building WMMA bf16 fragments
union Frag16 {
    v16bf          v;
    uint32_t       w[8];
    unsigned short u[16];
};

__device__ __forceinline__ unsigned short f2bf(float f) {
    // round-to-nearest-even f32 -> bf16
    uint32_t u = __float_as_uint(f);
    uint32_t r = (u + 0x7FFFu + ((u >> 16) & 1u)) >> 16;
    return (unsigned short)r;
}

// ---------------- weight convert + transpose ----------------
__global__ void cvt_w1_k(const float* __restrict__ W1, unsigned short* __restrict__ w1t) {
    int idx = blockIdx.x * blockDim.x + threadIdx.x;
    if (idx >= IN_C * HID_C) return;
    int k = idx / HID_C, n = idx % HID_C;
    w1t[(size_t)n * IN_C + k] = f2bf(W1[idx]);
}

__global__ void cvt_w2_k(const float* __restrict__ W2, unsigned short* __restrict__ w2t) {
    int idx = blockIdx.x * blockDim.x + threadIdx.x;
    if (idx >= HID_C * OUT_C) return;
    int k = idx / OUT_C, n = idx % OUT_C;
    w2t[(size_t)n * HID_C + k] = f2bf(W2[idx]);
}

// ---------------- GCN normalization ----------------
__global__ void deg_init_k(float* __restrict__ deg) {
    int i = blockIdx.x * blockDim.x + threadIdx.x;
    if (i < N_NODES) deg[i] = 1.0f;  // self-loop weight
}

__global__ void deg_accum_k(const int* __restrict__ ei, float* __restrict__ deg, int E) {
    int e = blockIdx.x * blockDim.x + threadIdx.x;
    if (e >= E) return;
    atomicAdd(&deg[ei[E + e]], 1.0f);  // segment_sum over col
}

__global__ void dinv_k(const float* __restrict__ deg, float* __restrict__ dinv) {
    int i = blockIdx.x * blockDim.x + threadIdx.x;
    if (i >= N_NODES) return;
    float d = deg[i];
    dinv[i] = (d > 0.0f) ? rsqrtf(d) : 0.0f;
}

__global__ void norm_k(const int* __restrict__ ei, const float* __restrict__ dinv,
                       float* __restrict__ norm, int E) {
    int e = blockIdx.x * blockDim.x + threadIdx.x;
    if (e >= E) return;
    norm[e] = dinv[ei[e]] * dinv[ei[E + e]];
}

// ---------------- GEMM1: h1 = relu(x @ W1 + b1), bf16 out ----------------
// One block (8 waves) = one 16-row M-strip x all 256 N columns.
// A tile (16x32 f32) staged to LDS as bf16 once per K-step; each wave does
// 2 N-tiles (2 wmma per K-step) off the shared A fragment.
#define A_ROW_US 40   // padded LDS row stride in ushorts (80B, 16B-aligned, bank-spread)
__global__ void __launch_bounds__(256) gemm1_k(const float* __restrict__ x,
                                               const unsigned short* __restrict__ w1t,
                                               const float* __restrict__ b1,
                                               unsigned short* __restrict__ h1b) {
    __shared__ __align__(16) unsigned short Atile[16 * A_ROW_US];
    int mt = blockIdx.x;                       // 3125 blocks
    int tid = threadIdx.x;
    int waveId = tid >> 5;
    int lane = tid & 31, ln = lane & 15, half = lane >> 4;
    int nt0 = waveId * 2, nt1 = nt0 + 1;
    int col0 = nt0 * 16 + ln, col1 = nt1 * 16 + ln;

    v8f c0 = {}, c1 = {};
    for (int kb = 0; kb < IN_C; kb += 32) {
        __syncthreads();  // protect previous iteration's LDS reads
        {   // cooperative load+convert: 16x32 f32 -> bf16 LDS, 2 elems/thread
            int e = tid * 2;
            int r = e >> 5, k = e & 31;
            const float* xp = x + (size_t)(mt * 16 + r) * IN_C + kb + k;
            float f0 = xp[0], f1 = xp[1];
            uint32_t packed = (uint32_t)f2bf(f0) | ((uint32_t)f2bf(f1) << 16);
            *(uint32_t*)&Atile[r * A_ROW_US + k] = packed;
        }
        __syncthreads();
        // A fragment from LDS: K runs [half*16,+16) of row ln
        Frag16 a;
        {
            const uint32_t* lp1 = (const uint32_t*)&Atile[ln * A_ROW_US + half * 16];
            const uint32_t* lp2 = (const uint32_t*)&Atile[ln * A_ROW_US + half * 16 + 8];
#pragma unroll
            for (int i = 0; i < 4; ++i) { a.w[i] = lp1[i]; a.w[4 + i] = lp2[i]; }
        }
        // B fragments (w1t is [N,K] bf16): 16 contiguous K at kb + half*16
        Frag16 b0, b1f;
        {
            const uint32_t* wr0 = (const uint32_t*)(w1t + (size_t)col0 * IN_C + kb + half * 16);
            const uint32_t* wr1 = (const uint32_t*)(w1t + (size_t)col1 * IN_C + kb + half * 16);
#pragma unroll
            for (int i = 0; i < 8; ++i) { b0.w[i] = wr0[i]; b1f.w[i] = wr1[i]; }
        }
        c0 = __builtin_amdgcn_wmma_f32_16x16x32_bf16(false, a.v, false, b0.v,
                                                     (short)0, c0, false, false);
        c1 = __builtin_amdgcn_wmma_f32_16x16x32_bf16(false, a.v, false, b1f.v,
                                                     (short)0, c1, false, false);
    }
#pragma unroll
    for (int r = 0; r < 8; ++r) {
        int m = mt * 16 + half * 8 + r;
        int n0 = nt0 * 16 + ln, n1 = nt1 * 16 + ln;
        float v0 = c0[r] + b1[n0];
        float v1 = c1[r] + b1[n1];
        h1b[(size_t)m * HID_C + n0] = f2bf(fmaxf(v0, 0.0f));
        h1b[(size_t)m * HID_C + n1] = f2bf(fmaxf(v1, 0.0f));
    }
}

// ---------------- GEMM2: h = h1 @ W2 + b2 ; hidden = coes[0]*h ----------------
// One wave = one 16-row M-tile x all 64 N columns (A frag reused 4x per K-step).
__global__ void gemm2_k(const unsigned short* __restrict__ h1b, const unsigned short* __restrict__ w2t,
                        const float* __restrict__ b2, const float* __restrict__ coes,
                        float* __restrict__ h, float* __restrict__ hidden) {
    int mt = (blockIdx.x * blockDim.x + threadIdx.x) >> 5;
    if (mt >= N_NODES / 16) return;
    int lane = threadIdx.x & 31;
    int ln = lane & 15, half = lane >> 4;
    int row = mt * 16 + ln;
    float coe0 = coes[0];

    v8f c[4] = {};
    Frag16 a, b;
#pragma unroll
    for (int kb = 0; kb < HID_C; kb += 32) {
        const uint32_t* hr1 = (const uint32_t*)(h1b + (size_t)row * HID_C + kb + half * 16);
        const uint32_t* hr2 = (const uint32_t*)(h1b + (size_t)row * HID_C + kb + half * 16 + 8);
#pragma unroll
        for (int i = 0; i < 4; ++i) { a.w[i] = hr1[i]; a.w[4 + i] = hr2[i]; }
#pragma unroll
        for (int nt = 0; nt < 4; ++nt) {
            int col = nt * 16 + ln;
            const uint32_t* wr = (const uint32_t*)(w2t + (size_t)col * HID_C + kb + half * 16);
#pragma unroll
            for (int i = 0; i < 8; ++i) b.w[i] = wr[i];
            c[nt] = __builtin_amdgcn_wmma_f32_16x16x32_bf16(false, a.v, false, b.v,
                                                            (short)0, c[nt], false, false);
        }
    }
#pragma unroll
    for (int nt = 0; nt < 4; ++nt) {
#pragma unroll
        for (int r = 0; r < 8; ++r) {
            int m = mt * 16 + half * 8 + r;
            int n = nt * 16 + ln;
            float val = c[nt][r] + b2[n];
            size_t idx = (size_t)m * OUT_C + n;
            h[idx] = val;
            hidden[idx] = coe0 * val;
        }
    }
}

// ---------------- propagation ----------------
__global__ void selfloop_init_k(float* __restrict__ hdst, const float* __restrict__ hsrc,
                                const float* __restrict__ dinv) {
    int idx = blockIdx.x * blockDim.x + threadIdx.x;
    if (idx >= N_NODES * OUT_C) return;
    int i = idx >> 6;
    float di = dinv[i];
    hdst[idx] = di * di * hsrc[idx];
}

// fused: hidden += coes[p]*hcur  AND  hnext = dinv^2 * hcur (self-loop init of next hop)
__global__ void accum_selfloop_k(float* __restrict__ hidden, const float* __restrict__ hcur,
                                 float* __restrict__ hnext, const float* __restrict__ dinv,
                                 const float* __restrict__ coes, int p) {
    int idx = blockIdx.x * blockDim.x + threadIdx.x;
    if (idx >= N_NODES * OUT_C) return;
    int i = idx >> 6;
    float v = hcur[idx];
    hidden[idx] += coes[p] * v;
    float di = dinv[i];
    hnext[idx] = di * di * v;
}

// 32 threads per edge, 2 features per thread (float2 gather, 2 atomics)
__global__ void scatter_k(float* __restrict__ hdst, const float* __restrict__ hsrc,
                          const int* __restrict__ ei, const float* __restrict__ norm, int E) {
    long long gid = (long long)blockIdx.x * blockDim.x + threadIdx.x;
    int e = (int)(gid >> 5);
    if (e >= E) return;
    int f = (int)(gid & 31) * 2;
    int r = ei[e], cdst = ei[E + e];
    float w = norm[e];
    float2 hv = *(const float2*)&hsrc[(size_t)r * OUT_C + f];
    float* dp = &hdst[(size_t)cdst * OUT_C + f];
    atomicAdd(dp,     w * hv.x);
    atomicAdd(dp + 1, w * hv.y);
}

__global__ void accum_k(float* __restrict__ hidden, const float* __restrict__ hcur,
                        const float* __restrict__ coes, int p1) {
    int idx = blockIdx.x * blockDim.x + threadIdx.x;
    if (idx >= N_NODES * OUT_C) return;
    hidden[idx] += coes[p1] * hcur[idx];
}

// ---------------- log_softmax: one wave per row of 64 ----------------
__global__ void logsoftmax_k(const float* __restrict__ hidden, float* __restrict__ out) {
    int wave = (blockIdx.x * blockDim.x + threadIdx.x) >> 5;
    if (wave >= N_NODES) return;
    int lane = threadIdx.x & 31;
    const float* rowp = hidden + (size_t)wave * OUT_C;
    float v0 = rowp[lane], v1 = rowp[lane + 32];
    float m = fmaxf(v0, v1);
#pragma unroll
    for (int o = 16; o > 0; o >>= 1) m = fmaxf(m, __shfl_xor(m, o));
    float s = expf(v0 - m) + expf(v1 - m);
#pragma unroll
    for (int o = 16; o > 0; o >>= 1) s += __shfl_xor(s, o);
    float lse = m + logf(s);
    out[(size_t)wave * OUT_C + lane]      = v0 - lse;
    out[(size_t)wave * OUT_C + lane + 32] = v1 - lse;
}

extern "C" void kernel_launch(void* const* d_in, const int* in_sizes, int n_in,
                              void* d_out, int out_size, void* d_ws, size_t ws_size,
                              hipStream_t stream) {
    const float* x    = (const float*)d_in[0];
    const int*   ei   = (const int*)d_in[1];
    const float* W1   = (const float*)d_in[2];
    const float* b1   = (const float*)d_in[3];
    const float* W2   = (const float*)d_in[4];
    const float* b2   = (const float*)d_in[5];
    const float* coes = (const float*)d_in[6];
    const int E = in_sizes[1] / 2;

    // workspace carve-out (256B aligned)
    char* ws = (char*)d_ws;
    size_t off = 0;
    auto carve = [&](size_t bytes) -> void* {
        void* p = ws + off;
        off = (off + bytes + 255) & ~(size_t)255;
        return p;
    };
    unsigned short* w1t   = (unsigned short*)carve((size_t)IN_C * HID_C * 2);
    unsigned short* w2t   = (unsigned short*)carve((size_t)HID_C * OUT_C * 2);
    unsigned short* h1b   = (unsigned short*)carve((size_t)N_NODES * HID_C * 2);
    float*          hA    = (float*)carve((size_t)N_NODES * OUT_C * 4);
    float*          hB    = (float*)carve((size_t)N_NODES * OUT_C * 4);
    float*          hidden= (float*)carve((size_t)N_NODES * OUT_C * 4);
    float*          deg   = (float*)carve((size_t)N_NODES * 4);
    float*          dinv  = (float*)carve((size_t)N_NODES * 4);
    float*          norm  = (float*)carve((size_t)E * 4);

    const int TB = 256;
    // weight prep
    cvt_w1_k<<<(IN_C * HID_C + TB - 1) / TB, TB, 0, stream>>>(W1, w1t);
    cvt_w2_k<<<(HID_C * OUT_C + TB - 1) / TB, TB, 0, stream>>>(W2, w2t);
    // gcn_norm
    deg_init_k<<<(N_NODES + TB - 1) / TB, TB, 0, stream>>>(deg);
    deg_accum_k<<<(E + TB - 1) / TB, TB, 0, stream>>>(ei, deg, E);
    dinv_k<<<(N_NODES + TB - 1) / TB, TB, 0, stream>>>(deg, dinv);
    norm_k<<<(E + TB - 1) / TB, TB, 0, stream>>>(ei, dinv, norm, E);
    // GEMMs on the matrix pipe
    gemm1_k<<<N_NODES / 16, TB, 0, stream>>>(x, w1t, b1, h1b);            // 1 block per M-strip
    gemm2_k<<<(N_NODES / 16 + 7) / 8, TB, 0, stream>>>(h1b, w2t, b2, coes, hA, hidden);
    // propagation hops:
    //   hop 1: init + scatter into hB from hA
    //   hops 2..P: fused (hidden += coes[p]*h^p ; init next dst) + scatter
    //   final: hidden += coes[P]*h^P
    const int NE = N_NODES * OUT_C;
    const unsigned int sc_blocks = (unsigned int)(((long long)E * 32 + TB - 1) / TB);
    float* hcur = hA;   // h^0
    float* hnxt = hB;
    selfloop_init_k<<<(NE + TB - 1) / TB, TB, 0, stream>>>(hnxt, hcur, dinv);
    scatter_k<<<sc_blocks, TB, 0, stream>>>(hnxt, hcur, ei, norm, E);      // hnxt = h^1
    for (int p = 1; p < P_HOPS; ++p) {
        // hnxt holds h^p; reuse hcur's buffer as next destination
        accum_selfloop_k<<<(NE + TB - 1) / TB, TB, 0, stream>>>(hidden, hnxt, hcur, dinv, coes, p);
        scatter_k<<<sc_blocks, TB, 0, stream>>>(hcur, hnxt, ei, norm, E);  // hcur = h^{p+1}
        float* t = hcur; hcur = hnxt; hnxt = t;                            // hnxt = h^{p+1}
    }
    accum_k<<<(NE + TB - 1) / TB, TB, 0, stream>>>(hidden, hnxt, coes, P_HOPS);
    // output
    logsoftmax_k<<<(N_NODES + 7) / 8, TB, 0, stream>>>(hidden, (float*)d_out);
}
